// EdgeHeatmapGenerator_50448685859365
// MI455X (gfx1250) — compile-verified, matrix operand order checked by reading, not run on hardware
//
#include <hip/hip_runtime.h>
#include <hip/hip_bf16.h>

typedef __attribute__((ext_vector_type(16))) _Float16 v16h;
typedef __attribute__((ext_vector_type(8)))  _Float16 v8h;
typedef __attribute__((ext_vector_type(8)))  float    v8f;

#define D_DIM   128      // feature dim
#define LDW     136      // LDS row stride in halves (272B = 17*16B, conflict-free, 16B aligned)
#define TILE_E  128      // edges per workgroup
#define NTHREADS 256     // 8 wave32

// ---------------------------------------------------------------------------
// Fill kernel: heatmap = log(0 + EPS) everywhere
// ---------------------------------------------------------------------------
__global__ void heatmap_fill_kernel(float* __restrict__ out, size_t n, float v) {
    size_t i = (size_t)blockIdx.x * blockDim.x + threadIdx.x;
    size_t stride = (size_t)gridDim.x * blockDim.x;
    for (; i < n; i += stride) out[i] = v;
}

// ---------------------------------------------------------------------------
// Load a 16x32 f16 fragment (A or B operand) from LDS in the CDNA5 WMMA
// register layout (ISA 7.12.2, 16-bit A-matrix 16x32):
//   lanes 0-15 : row = rowbase+lane,    halves 0-7 = K[kb..kb+7],  8-15 = K[kb+16..kb+23]
//   lanes 16-31: row = rowbase+lane-16, halves 0-7 = K[kb+8..+15], 8-15 = K[kb+24..+31]
// Two 16B ds_load_b128 per lane.
// ---------------------------------------------------------------------------
__device__ __forceinline__ v16h load_frag_16x32(const _Float16* base, int rowbase,
                                                int kbase, int lane) {
    const int hsel = lane >> 4;                 // 0 or 1
    const int row  = rowbase + (lane & 15);
    const _Float16* rp = base + row * LDW;
    v8h lo = *(const v8h*)(rp + kbase + hsel * 8);
    v8h hi = *(const v8h*)(rp + kbase + 16 + hsel * 8);
    v16h out;
#pragma unroll
    for (int i = 0; i < 8; ++i) { out[i] = lo[i]; out[i + 8] = hi[i]; }
    return out;
}

__device__ __forceinline__ float silu_f(float x) {
    return x / (1.0f + __expf(-x));
}

// ---------------------------------------------------------------------------
// One tile: 128 edges. X(128x128) -> silu(X W1^T + b1) -> silu(. W2^T + b2)
// -> sigmoid(. Wout^T + bout) -> scatter log(val+EPS)
// ---------------------------------------------------------------------------
__global__ __launch_bounds__(NTHREADS, 1)
void edge_mlp_scatter_kernel(const float* __restrict__ edge_attr,
                             const long long* __restrict__ edge_index,
                             const float* __restrict__ W1, const float* __restrict__ b1,
                             const float* __restrict__ W2, const float* __restrict__ b2,
                             const float* __restrict__ Wout, const float* __restrict__ bout,
                             float* __restrict__ out,
                             int E, int N) {
    __shared__ __align__(16) _Float16 Xs[TILE_E * LDW];  // X tile, later H2
    __shared__ __align__(16) _Float16 Hs[TILE_E * LDW];  // H1
    __shared__ __align__(16) _Float16 Ws[D_DIM * LDW];   // W1 then W2 (f16)
    __shared__ float b1s[D_DIM];
    __shared__ float b2s[D_DIM];
    __shared__ float wouts[D_DIM];

    const int tid  = threadIdx.x;
    const int lane = tid & 31;
    const int wave = tid >> 5;
    const int rowbase = wave * 16;

    const long long ge0 = (long long)blockIdx.x * TILE_E;  // first global edge

    // ---- stage X tile (f32 -> f16) : 128x128 floats, float4 vectorized ----
    {
        const float4* src = (const float4*)(edge_attr + ge0 * D_DIM);
#pragma unroll 4
        for (int i = tid; i < (TILE_E * D_DIM / 4); i += NTHREADS) {
            float4 v = src[i];
            int row = (i * 4) / D_DIM;
            int col = (i * 4) % D_DIM;
            _Float16* d = &Xs[row * LDW + col];
            d[0] = (_Float16)v.x; d[1] = (_Float16)v.y;
            d[2] = (_Float16)v.z; d[3] = (_Float16)v.w;
        }
    }
    // ---- stage W1 (f32 -> f16) + biases + wout ----
    {
        const float4* src = (const float4*)W1;
#pragma unroll 4
        for (int i = tid; i < (D_DIM * D_DIM / 4); i += NTHREADS) {
            float4 v = src[i];
            int row = (i * 4) / D_DIM;
            int col = (i * 4) % D_DIM;
            _Float16* d = &Ws[row * LDW + col];
            d[0] = (_Float16)v.x; d[1] = (_Float16)v.y;
            d[2] = (_Float16)v.z; d[3] = (_Float16)v.w;
        }
        if (tid < D_DIM) { b1s[tid] = b1[tid]; b2s[tid] = b2[tid]; wouts[tid] = Wout[tid]; }
    }
    __syncthreads();

    // ================= layer 1: Hs = silu(Xs @ W1^T + b1) =================
    {
        v16h a0 = load_frag_16x32(Xs, rowbase,  0, lane);
        v16h a1 = load_frag_16x32(Xs, rowbase, 32, lane);
        v16h a2 = load_frag_16x32(Xs, rowbase, 64, lane);
        v16h a3 = load_frag_16x32(Xs, rowbase, 96, lane);
#pragma unroll
        for (int n = 0; n < 8; ++n) {
            // B[k][f] = W1[f][k] : same fragment layout as A with row = output col
            v16h bfr0 = load_frag_16x32(Ws, n * 16,  0, lane);
            v16h bfr1 = load_frag_16x32(Ws, n * 16, 32, lane);
            v16h bfr2 = load_frag_16x32(Ws, n * 16, 64, lane);
            v16h bfr3 = load_frag_16x32(Ws, n * 16, 96, lane);
            v8f acc = {};
            acc = __builtin_amdgcn_wmma_f32_16x16x32_f16(false, a0, false, bfr0, (short)0, acc, false, false);
            acc = __builtin_amdgcn_wmma_f32_16x16x32_f16(false, a1, false, bfr1, (short)0, acc, false, false);
            acc = __builtin_amdgcn_wmma_f32_16x16x32_f16(false, a2, false, bfr2, (short)0, acc, false, false);
            acc = __builtin_amdgcn_wmma_f32_16x16x32_f16(false, a3, false, bfr3, (short)0, acc, false, false);
            // epilogue: bias + silu; C/D layout: VGPR j -> M = j + 8*(lane>=16), N = lane&15
            const int col  = n * 16 + (lane & 15);
            const int roff = (lane >> 4) * 8;
            const float bias = b1s[col];
#pragma unroll
            for (int j = 0; j < 8; ++j) {
                float s = silu_f(acc[j] + bias);
                Hs[(rowbase + roff + j) * LDW + col] = (_Float16)s;
            }
        }
    }
    __syncthreads();

    // ---- reload Ws with W2 ----
    {
        const float4* src = (const float4*)W2;
#pragma unroll 4
        for (int i = tid; i < (D_DIM * D_DIM / 4); i += NTHREADS) {
            float4 v = src[i];
            int row = (i * 4) / D_DIM;
            int col = (i * 4) % D_DIM;
            _Float16* d = &Ws[row * LDW + col];
            d[0] = (_Float16)v.x; d[1] = (_Float16)v.y;
            d[2] = (_Float16)v.z; d[3] = (_Float16)v.w;
        }
    }
    __syncthreads();

    // ================= layer 2: Xs = silu(Hs @ W2^T + b2) =================
    {
        v16h a0 = load_frag_16x32(Hs, rowbase,  0, lane);
        v16h a1 = load_frag_16x32(Hs, rowbase, 32, lane);
        v16h a2 = load_frag_16x32(Hs, rowbase, 64, lane);
        v16h a3 = load_frag_16x32(Hs, rowbase, 96, lane);
#pragma unroll
        for (int n = 0; n < 8; ++n) {
            v16h bfr0 = load_frag_16x32(Ws, n * 16,  0, lane);
            v16h bfr1 = load_frag_16x32(Ws, n * 16, 32, lane);
            v16h bfr2 = load_frag_16x32(Ws, n * 16, 64, lane);
            v16h bfr3 = load_frag_16x32(Ws, n * 16, 96, lane);
            v8f acc = {};
            acc = __builtin_amdgcn_wmma_f32_16x16x32_f16(false, a0, false, bfr0, (short)0, acc, false, false);
            acc = __builtin_amdgcn_wmma_f32_16x16x32_f16(false, a1, false, bfr1, (short)0, acc, false, false);
            acc = __builtin_amdgcn_wmma_f32_16x16x32_f16(false, a2, false, bfr2, (short)0, acc, false, false);
            acc = __builtin_amdgcn_wmma_f32_16x16x32_f16(false, a3, false, bfr3, (short)0, acc, false, false);
            const int col  = n * 16 + (lane & 15);
            const int roff = (lane >> 4) * 8;
            const float bias = b2s[col];
#pragma unroll
            for (int j = 0; j < 8; ++j) {
                float s = silu_f(acc[j] + bias);
                Xs[(rowbase + roff + j) * LDW + col] = (_Float16)s;
            }
        }
    }
    __syncthreads();

    // ========== head: sigmoid(H2 . wout + bout), scatter log(val+EPS) =====
    if (tid < TILE_E) {
        float dot = bout[0];
        const _Float16* hrow = &Xs[tid * LDW];
#pragma unroll 8
        for (int d = 0; d < D_DIM; ++d) dot += (float)hrow[d] * wouts[d];
        float val = 1.0f / (1.0f + __expf(-dot));
        float r = __logf(val + 1e-10f);

        const int b = (int)(ge0 / E);        // TILE_E divides E: no straddle
        const int e = (int)(ge0 % E) + tid;
        long long s = edge_index[((long long)b * 2 + 0) * E + e];
        long long t = edge_index[((long long)b * 2 + 1) * E + e];
        out[(size_t)b * N * N + (size_t)s * N + (size_t)t] = r;
    }
}

// ---------------------------------------------------------------------------
extern "C" void kernel_launch(void* const* d_in, const int* in_sizes, int n_in,
                              void* d_out, int out_size, void* d_ws, size_t ws_size,
                              hipStream_t stream) {
    const float*     edge_attr  = (const float*)d_in[0];
    const long long* edge_index = (const long long*)d_in[1];
    const float*     W1   = (const float*)d_in[2];
    const float*     b1   = (const float*)d_in[3];
    const float*     W2   = (const float*)d_in[4];
    const float*     b2   = (const float*)d_in[5];
    const float*     Wout = (const float*)d_in[6];
    const float*     bout = (const float*)d_in[7];
    float*           out  = (float*)d_out;

    const int E = 32000;   // edges per batch (reference)
    const int N = 1000;    // num_nodes (reference)

    // 1) fill heatmap with log(EPS)
    const float logeps = -23.025850929940457f;  // ln(1e-10)
    heatmap_fill_kernel<<<4096, NTHREADS, 0, stream>>>(out, (size_t)out_size, logeps);

    // 2) edge MLP + scatter; tiles = total_edges / TILE_E
    const long long total_edges = (long long)in_sizes[0] / D_DIM;   // B*E
    const int tiles = (int)(total_edges / TILE_E);                  // 8000
    edge_mlp_scatter_kernel<<<tiles, NTHREADS, 0, stream>>>(
        edge_attr, edge_index, W1, b1, W2, b2, Wout, bout, out, E, N);
}
